// HyperCategoryFramework_54434415509945
// MI455X (gfx1250) — matrix operand order repaired and checked by reading, not compiled, not run on hardware
//
#include <hip/hip_runtime.h>
#include <hip/hip_bf16.h>
#include <math.h>

typedef __attribute__((ext_vector_type(16))) __bf16 v16bf;
typedef __attribute__((ext_vector_type(8)))  float  v8f;

#define N_FILT 28
#define WDIM   144
#define WKPAD  160           // K padded to multiple of 32 for guard-free WMMA loops
#define CDIM   256
#define NHEAD  4
#define DHEAD  64
#define DFFN   2048
#define NCLS   10
#define BATCH  32
#define NPAIR  378
#define NTRI   3276
#define MTOK   3682          // 28 + 378 + 3276
#define MPAD   3696          // 231 * 16
#define EPSC   17.0f
#define SHARPC 10.0f

__device__ __forceinline__ float sigmoidf_(float z){ return 1.0f/(1.0f+__expf(-z)); }
__device__ __forceinline__ float geluf_(float x){ return 0.5f*x*(1.0f+erff(x*0.70710678118654752f)); }

// pack two contiguous float4s into 8 consecutive bf16 vector elements
__device__ __forceinline__ void pack8(v16bf& v, int base, float4 a, float4 b){
    v[base+0]=(__bf16)a.x; v[base+1]=(__bf16)a.y; v[base+2]=(__bf16)a.z; v[base+3]=(__bf16)a.w;
    v[base+4]=(__bf16)b.x; v[base+5]=(__bf16)b.y; v[base+6]=(__bf16)b.z; v[base+7]=(__bf16)b.w;
}
// A-fragment (16x32 bf16): per-lane one contiguous 16-float run at p
__device__ __forceinline__ v16bf load_afrag(const float* p){
    const float4* q0 = (const float4*)p;
    const float4* q1 = (const float4*)(p + 8);
    float4 a0 = q0[0], a1 = q0[1], a2 = q1[0], a3 = q1[1];
    v16bf v;
    pack8(v, 0, a0, a1);
    pack8(v, 8, a2, a3);
    return v;
}
// B-fragment (32x16 bf16): per-lane one contiguous 16-float run
__device__ __forceinline__ v16bf load_bfrag16(const float* p){
    const float4* q = (const float4*)p;
    float4 b0 = q[0], b1 = q[1], b2 = q[2], b3 = q[3];
    v16bf v;
    pack8(v, 0, b0, b1);
    pack8(v, 8, b2, b3);
    return v;
}

// ---------------------------------------------------------------------------
// Conv(valid 3x3) + bias + relu + spatial mean  ->  y1[B][28]
// ---------------------------------------------------------------------------
__global__ void k_conv_mean(const float* __restrict__ x, const float* __restrict__ cw,
                            const float* __restrict__ cb, float* __restrict__ y1)
{
    __shared__ float wsm[WDIM];
    __shared__ float red[256];
    int b = blockIdx.x / N_FILT, n = blockIdx.x % N_FILT;
    if (threadIdx.x < WDIM) wsm[threadIdx.x] = cw[n*WDIM + threadIdx.x];
    __syncthreads();
    float bias = cb[n];
    float acc = 0.f;
    const float* xb = x + (size_t)b*16*1024;
    for (int p = threadIdx.x; p < 900; p += 256) {
        int r = p/30, c = p%30;
        float s = bias;
        #pragma unroll
        for (int ci = 0; ci < 16; ++ci)
            #pragma unroll
            for (int kh = 0; kh < 3; ++kh)
                #pragma unroll
                for (int kw = 0; kw < 3; ++kw)
                    s += xb[ci*1024 + (r+kh)*32 + (c+kw)] * wsm[ci*9 + kh*3 + kw];
        acc += fmaxf(s, 0.f);
    }
    red[threadIdx.x] = acc;
    __syncthreads();
    for (int st = 128; st > 0; st >>= 1){
        if (threadIdx.x < st) red[threadIdx.x] += red[threadIdx.x+st];
        __syncthreads();
    }
    if (threadIdx.x == 0) y1[b*N_FILT + n] = red[0] * (1.f/900.f);
}

__global__ void k_fc(const float* __restrict__ y1, const float* __restrict__ fw,
                     const float* __restrict__ fb, float* __restrict__ y)
{
    int t = threadIdx.x; if (t >= BATCH*NCLS) return;
    int b = t/NCLS, c = t%NCLS;
    float s = fb[c];
    for (int n = 0; n < N_FILT; ++n) s += y1[b*N_FILT+n]*fw[c*N_FILT+n];
    y[t] = s;
}

// ---------------------------------------------------------------------------
// hypergraph pooled-feature builder (rows padded to WKPAD cols, zero-filled)
// ---------------------------------------------------------------------------
__global__ void k_base_rows(const float* __restrict__ cw, float* __restrict__ pooled,
                            float* __restrict__ orders, float* __restrict__ wts)
{
    int blk = blockIdx.x, t = threadIdx.x;
    if (blk < N_FILT) {
        if (t < WKPAD)
            pooled[(size_t)blk*WKPAD + t] = (t < WDIM) ? cw[blk*WDIM + t] * (1.f/(1.f+1e-6f)) : 0.f;
        if (t == 0){ orders[blk] = 1.f; wts[blk] = 1.f; }
    } else {
        int row = MTOK + (blk - N_FILT);          // zero padding rows
        if (row < MPAD){
            if (t < WKPAD) pooled[(size_t)row*WKPAD + t] = 0.f;
            if (t == 0){ orders[row] = 0.f; wts[row] = 0.f; }
        }
    }
}

// zero-padded copy of lin_in_w: [CDIM][WDIM] -> [CDIM][WKPAD]
__global__ void k_pad_linin(const float* __restrict__ w, float* __restrict__ wp)
{
    int r = blockIdx.x, t = threadIdx.x;
    if (t < WKPAD) wp[(size_t)r*WKPAD + t] = (t < WDIM) ? w[r*WDIM + t] : 0.f;
}

__global__ void k_pairs(const float* __restrict__ cw, float* __restrict__ pooled,
                        float* __restrict__ orders, float* __restrict__ wts,
                        float* __restrict__ Dm)
{
    __shared__ float red[256];
    int p = blockIdx.x, t = threadIdx.x;
    int i = 0, rem = p;
    for (i = 0; i < N_FILT-1; ++i){ int cnt = N_FILT-1-i; if (rem < cnt) break; rem -= cnt; }
    int j = i + 1 + rem;
    float wi = (t < WDIM) ? cw[i*WDIM+t] : 0.f;
    float wj = (t < WDIM) ? cw[j*WDIM+t] : 0.f;
    float df = wi - wj;
    red[t] = df*df;
    __syncthreads();
    for (int st = 128; st > 0; st >>= 1){
        if (t < st) red[t] += red[t+st];
        __syncthreads();
    }
    float d   = sqrtf(red[0]);
    float val = sigmoidf_(-SHARPC*(d - EPSC));
    int row = N_FILT + p;
    if (t < WKPAD)
        pooled[(size_t)row*WKPAD + t] = (t < WDIM) ? val*(wi+wj)/(2.f*val + 1e-6f) : 0.f;
    if (t == 0){
        orders[row] = 2.f*val;
        wts[row]    = (val > 1e-6f) ? 1.f : 0.f;
        Dm[i*N_FILT+j] = d; Dm[j*N_FILT+i] = d;
    }
}

__global__ void k_triples(const float* __restrict__ cw, const float* __restrict__ Dm,
                          float* __restrict__ pooled, float* __restrict__ orders,
                          float* __restrict__ wts)
{
    int p = blockIdx.x, t = threadIdx.x;
    int rem = p, i, j, k;
    for (i = 0; i < N_FILT; ++i){ int c2 = (N_FILT-1-i)*(N_FILT-2-i)/2; if (rem < c2) break; rem -= c2; }
    for (j = i+1; j < N_FILT; ++j){ int cnt = N_FILT-1-j; if (rem < cnt) break; rem -= cnt; }
    k = j + 1 + rem;
    int row = N_FILT + NPAIR + p;
    if (t < WKPAD)
        pooled[(size_t)row*WKPAD + t] = (t < WDIM)
            ? (cw[i*WDIM+t] + cw[j*WDIM+t] + cw[k*WDIM+t]) * (1.f/(3.f+1e-6f)) : 0.f;
    if (t == 0){
        float w = (Dm[i*N_FILT+j] <= EPSC && Dm[j*N_FILT+k] <= EPSC && Dm[i*N_FILT+k] <= EPSC) ? 1.f : 0.f;
        orders[row] = 3.f; wts[row] = w;
    }
}

// V transpose: vt[head*64+ch][key] = qkv[key][2*CDIM + head*64 + ch]
// makes the P@V B-fragments contiguous (4 x b128 instead of 16 strided dwords)
__global__ void k_vtrans(const float* __restrict__ qkv, float* __restrict__ vt)
{
    int key = blockIdx.x, t = threadIdx.x;       // t = head*64+ch in [0,256)
    vt[(size_t)t*MPAD + key] = qkv[(size_t)key*(3*CDIM) + 2*CDIM + t];
}

// ---------------------------------------------------------------------------
// generic WMMA GEMM:  C[M,N] = act( A[M,K] x op(W) + bias + Cadd )
//   transw=1 : W is [N,K] row-major      transw=0 : W is [K,N]
// K must be a multiple of 32 (guard-free b128 fragment loads).
// block = 128 threads = 4 waves; wave w owns C tile (blockIdx.x*16, (4*by+w)*16)
// ---------------------------------------------------------------------------
__global__ void k_gemm(const float* __restrict__ A, const float* __restrict__ Wm,
                       const float* __restrict__ bias, const float* __restrict__ Cadd,
                       float* __restrict__ C, int Mn, int Nn, int Kn, int transw, int act)
{
    int wave = threadIdx.x >> 5;
    int lane = threadIdx.x & 31;
    int mrow = lane & 15, hi = lane >> 4;
    int m0 = blockIdx.x * 16;
    int n0 = (blockIdx.y * 4 + wave) * 16;
    if (m0 >= Mn || n0 >= Nn) return;

    v8f acc = {};
    const float* Arow = A + (size_t)(m0 + mrow) * Kn + hi*16;
    if (transw) {
        const float* Wrow = Wm + (size_t)(n0 + mrow) * Kn + hi*16;
        for (int k0 = 0; k0 < Kn; k0 += 32) {
            __builtin_prefetch((const void*)(Arow + k0 + 128), 0, 1);
            v16bf av = load_afrag(Arow + k0);
            v16bf bv = load_bfrag16(Wrow + k0);
            acc = __builtin_amdgcn_wmma_f32_16x16x32_bf16(false, av, false, bv,
                                                          (short)0, acc, false, false);
        }
    } else {
        const float* Wcol = Wm + (size_t)(hi*16)*Nn + n0 + mrow;
        for (int k0 = 0; k0 < Kn; k0 += 32) {
            v16bf av = load_afrag(Arow + k0);
            v16bf bv;
            #pragma unroll
            for (int e = 0; e < 16; ++e)
                bv[e] = (__bf16)Wcol[(size_t)(k0 + e)*Nn];
            acc = __builtin_amdgcn_wmma_f32_16x16x32_bf16(false, av, false, bv,
                                                          (short)0, acc, false, false);
        }
    }
    int n = n0 + mrow;
    float bval = bias ? bias[n] : 0.f;
    #pragma unroll
    for (int r = 0; r < 8; ++r) {                  // C elem r -> M = m0 + r + hi*8
        int m = m0 + r + hi*8;
        float v = acc[r] + bval;
        if (Cadd) v += Cadd[(size_t)m*Nn + n];
        if (act == 1)      v = fmaxf(v, 0.f);
        else if (act == 2) v = geluf_(v);
        C[(size_t)m*Nn + n] = v;
    }
}

// ---------------------------------------------------------------------------
// fused flash attention: block = 16-query tile, wave = head (NHEAD==4 waves)
// K fragments from qkv (contiguous), V fragments from transposed vt (contiguous)
// ---------------------------------------------------------------------------
__global__ void k_attn(const float* __restrict__ qkv, const float* __restrict__ vt,
                       float* __restrict__ outp)
{
    __shared__ float pls[4*16*32];
    int wave = threadIdx.x >> 5;                   // = head
    int lane = threadIdx.x & 31;
    int mrow = lane & 15, hi = lane >> 4;
    int head = wave;
    int m0   = blockIdx.x * 16;
    float* P = pls + wave*512;                     // 16x32 probability scratch

    // Q fragments for dh chunks [0,32) and [32,64)
    const float* qrow = qkv + (size_t)(m0 + mrow)*(3*CDIM) + head*DHEAD + hi*16;
    v16bf aq0 = load_afrag(qrow);
    v16bf aq1 = load_afrag(qrow + 32);

    // per-lane V^T row base: channel = head*64 + ct*16 + mrow, key-major
    const float* vrow = vt + (size_t)(head*DHEAD + mrow)*MPAD + hi*16;

    v8f ov[4] = {};                                // O tile: 16 x 64 (4 n-tiles)
    float lsum[8] = {};                            // per-lane partial row sums

    for (int j0 = 0; j0 < MPAD; j0 += 32) {
        // --- scores for 32 keys (two 16-key tiles) ---
        #pragma unroll
        for (int tk = 0; tk < 2; ++tk) {
            int key = j0 + tk*16 + mrow;
            const float* krow = qkv + (size_t)key*(3*CDIM) + CDIM + head*DHEAD + hi*16;
            v16bf b0 = load_bfrag16(krow);
            v16bf b1 = load_bfrag16(krow + 32);
            v8f s = {};
            s = __builtin_amdgcn_wmma_f32_16x16x32_bf16(false, aq0, false, b0, (short)0, s, false, false);
            s = __builtin_amdgcn_wmma_f32_16x16x32_bf16(false, aq1, false, b1, (short)0, s, false, false);
            float mask = (key < MTOK) ? 1.f : 0.f; // kill padded keys
            #pragma unroll
            for (int r = 0; r < 8; ++r){
                float pv = mask * __expf(s[r] * 0.125f);   // scale = 1/sqrt(64)
                lsum[r] += pv;
                P[(r + hi*8)*32 + tk*16 + mrow] = pv;      // C-layout -> row-major LDS
            }
        }
        // --- reload P in A-layout (16x32) ---
        v16bf pa;
        #pragma unroll
        for (int e = 0; e < 16; ++e)
            pa[e] = (__bf16)P[mrow*32 + hi*16 + e];
        // --- O += P @ V(32 keys x 64 ch), 4 channel tiles, contiguous b128 loads ---
        #pragma unroll
        for (int ct = 0; ct < 4; ++ct) {
            v16bf bv = load_bfrag16(vrow + (size_t)(ct*16)*MPAD + j0);
            ov[ct] = __builtin_amdgcn_wmma_f32_16x16x32_bf16(false, pa, false, bv,
                                                             (short)0, ov[ct], false, false);
        }
    }

    // reduce row sums across the 16 lanes holding the same rows (xor bits 0..3)
    #pragma unroll
    for (int r = 0; r < 8; ++r){
        float s = lsum[r];
        s += __shfl_xor(s, 1); s += __shfl_xor(s, 2);
        s += __shfl_xor(s, 4); s += __shfl_xor(s, 8);
        lsum[r] = 1.f / s;
    }
    #pragma unroll
    for (int ct = 0; ct < 4; ++ct)
        #pragma unroll
        for (int r = 0; r < 8; ++r)
            outp[(size_t)(m0 + r + hi*8)*CDIM + head*DHEAD + ct*16 + mrow] = ov[ct][r] * lsum[r];
}

// ---------------------------------------------------------------------------
// residual + layernorm (one row per block, 256 threads = 1/col)
// ---------------------------------------------------------------------------
__global__ void k_ln(const float* X, const float* R, const float* g, const float* bb, float* outp)
{
    __shared__ float red[256];
    int row = blockIdx.x, t = threadIdx.x;
    float v = X[(size_t)row*CDIM + t] + R[(size_t)row*CDIM + t];
    red[t] = v; __syncthreads();
    for (int st = 128; st > 0; st >>= 1){ if (t < st) red[t] += red[t+st]; __syncthreads(); }
    float mean = red[0] * (1.f/CDIM);
    __syncthreads();
    float d = v - mean;
    red[t] = d*d; __syncthreads();
    for (int st = 128; st > 0; st >>= 1){ if (t < st) red[t] += red[t+st]; __syncthreads(); }
    float var = red[0] * (1.f/CDIM);
    outp[(size_t)row*CDIM + t] = d * rsqrtf(var + 1e-5f) * g[t] + bb[t];
}

// ---------------------------------------------------------------------------
// weighted mean of H = is_pair*transformed + (1-is_pair)*contexts
// ---------------------------------------------------------------------------
__global__ void k_hmean(const float* __restrict__ ctx, const float* __restrict__ tr,
                        const float* __restrict__ orders, const float* __restrict__ wts,
                        float* __restrict__ hm)
{
    __shared__ float s1[256], s2[256];
    int col = blockIdx.x, t = threadIdx.x;
    float s = 0.f, w = 0.f;
    for (int r = t; r < MTOK; r += 256) {
        float wr = wts[r];
        float ip = sigmoidf_((orders[r] - 2.f) * 10.f);
        float c  = ctx[(size_t)r*CDIM + col];
        float h  = ip*tr[(size_t)r*CDIM + col] + (1.f - ip)*c;
        s += wr*h; w += wr;
    }
    s1[t] = s; s2[t] = w; __syncthreads();
    for (int st = 128; st > 0; st >>= 1){
        if (t < st){ s1[t] += s1[t+st]; s2[t] += s2[t+st]; }
        __syncthreads();
    }
    if (t == 0) hm[col] = s1[0] / s2[0];
}

__global__ void k_proj(const float* __restrict__ hm, const float* __restrict__ omega,
                       float* __restrict__ proj)
{
    int i = threadIdx.x;
    float s = 0.f;
    for (int j = 0; j < CDIM; ++j) s += hm[j]*omega[(size_t)i*CDIM + j];
    proj[i] = s;
}

__global__ void k_final(const float* __restrict__ y, const float* __restrict__ proj,
                        const float* __restrict__ caw, const float* __restrict__ cab,
                        float* __restrict__ outp)
{
    int t = threadIdx.x; if (t >= BATCH*NCLS) return;
    int c = t % NCLS;
    float s = cab[c];
    for (int i = 0; i < CDIM; ++i) s += proj[i]*caw[(size_t)c*CDIM + i];
    outp[t] = y[t] + s;
}

// ---------------------------------------------------------------------------
extern "C" void kernel_launch(void* const* d_in, const int* in_sizes, int n_in,
                              void* d_out, int out_size, void* d_ws, size_t ws_size,
                              hipStream_t stream)
{
    (void)in_sizes; (void)n_in; (void)out_size; (void)ws_size;
    const float* x         = (const float*)d_in[0];
    const float* conv_w    = (const float*)d_in[1];
    const float* conv_b    = (const float*)d_in[2];
    const float* fc_w      = (const float*)d_in[3];
    const float* fc_b      = (const float*)d_in[4];
    const float* lin_in_w  = (const float*)d_in[5];
    const float* lin_in_b  = (const float*)d_in[6];
    const float* in_proj_w = (const float*)d_in[7];
    const float* in_proj_b = (const float*)d_in[8];
    const float* out_proj_w= (const float*)d_in[9];
    const float* out_proj_b= (const float*)d_in[10];
    const float* ln1_g     = (const float*)d_in[11];
    const float* ln1_b     = (const float*)d_in[12];
    const float* ffn_w1    = (const float*)d_in[13];
    const float* ffn_b1    = (const float*)d_in[14];
    const float* ffn_w2    = (const float*)d_in[15];
    const float* ffn_b2    = (const float*)d_in[16];
    const float* ln2_g     = (const float*)d_in[17];
    const float* ln2_b     = (const float*)d_in[18];
    const float* lin_out_w = (const float*)d_in[19];
    const float* lin_out_b = (const float*)d_in[20];
    const float* omega     = (const float*)d_in[21];
    const float* lam_w     = (const float*)d_in[22];
    const float* lam_b     = (const float*)d_in[23];
    const float* ca_w      = (const float*)d_in[24];
    const float* ca_b      = (const float*)d_in[25];
    float* outp = (float*)d_out;

    float* ws = (float*)d_ws;
    size_t off = 0;
    float* ybuf   = ws + off; off += 512;
    float* y1     = ws + off; off += 1024;
    float* Dm     = ws + off; off += 1024;
    float* orders = ws + off; off += MPAD;
    float* wts    = ws + off; off += MPAD;
    float* linwp  = ws + off; off += (size_t)CDIM*WKPAD;     // padded lin_in_w
    float* pooled = ws + off; off += (size_t)MPAD*WKPAD;
    float* hbuf   = ws + off; off += (size_t)MPAD*CDIM;
    float* qkvbuf = ws + off; off += (size_t)MPAD*3*CDIM;
    float* vtbuf  = ws + off; off += (size_t)MPAD*CDIM;      // transposed V
    float* abuf   = ws + off; off += (size_t)MPAD*CDIM;
    float* obuf   = ws + off; off += (size_t)MPAD*CDIM;
    float* ctxbuf = ws + off; off += (size_t)MPAD*CDIM;
    float* ffbuf  = ws + off; off += (size_t)MPAD*DFFN;
    float* hmeanb = ws + off; off += 256;
    float* projb  = ws + off; off += 256;
    float* t1 = abuf;    // dead after out_proj
    float* t2 = ffbuf;   // dead after FFN

    const int gM = MPAD/16;   // 231 m-tiles

    // classifier head
    k_conv_mean<<<BATCH*N_FILT, 256, 0, stream>>>(x, conv_w, conv_b, y1);
    k_fc<<<1, 352, 0, stream>>>(y1, fc_w, fc_b, ybuf);
    // hypergraph token builder
    k_base_rows<<<N_FILT + (MPAD - MTOK), 256, 0, stream>>>(conv_w, pooled, orders, wts);
    k_pad_linin<<<CDIM, 160, 0, stream>>>(lin_in_w, linwp);
    k_pairs<<<NPAIR, 256, 0, stream>>>(conv_w, pooled, orders, wts, Dm);
    k_triples<<<NTRI, 256, 0, stream>>>(conv_w, Dm, pooled, orders, wts);
    // transformer block (all WMMA)
    k_gemm<<<dim3(gM, 4),        128, 0, stream>>>(pooled, linwp, lin_in_b, nullptr, hbuf,   MPAD, CDIM,   WKPAD, 1, 0);
    k_gemm<<<dim3(gM, 12),       128, 0, stream>>>(hbuf,   in_proj_w, in_proj_b, nullptr, qkvbuf, MPAD, 3*CDIM, CDIM, 1, 0);
    k_vtrans<<<MPAD, 256, 0, stream>>>(qkvbuf, vtbuf);
    k_attn<<<gM, 128, 0, stream>>>(qkvbuf, vtbuf, abuf);
    k_gemm<<<dim3(gM, 4),        128, 0, stream>>>(abuf,   out_proj_w, out_proj_b, nullptr, obuf, MPAD, CDIM, CDIM, 1, 0);
    k_ln<<<MPAD, 256, 0, stream>>>(hbuf, obuf, ln1_g, ln1_b, hbuf);
    k_gemm<<<dim3(gM, DFFN/64),  128, 0, stream>>>(hbuf,   ffn_w1, ffn_b1, nullptr, ffbuf, MPAD, DFFN, CDIM, 1, 1);
    k_gemm<<<dim3(gM, 4),        128, 0, stream>>>(ffbuf,  ffn_w2, ffn_b2, nullptr, obuf,  MPAD, CDIM, DFFN, 1, 0);
    k_ln<<<MPAD, 256, 0, stream>>>(hbuf, obuf, ln2_g, ln2_b, hbuf);
    // epilogue projections
    k_gemm<<<dim3(gM, 4), 128, 0, stream>>>(hbuf,   lin_out_w, lin_out_b, nullptr, ctxbuf, MPAD, CDIM, CDIM, 1, 0);
    k_gemm<<<dim3(gM, 4), 128, 0, stream>>>(ctxbuf, omega,     nullptr,   nullptr, t1,     MPAD, CDIM, CDIM, 0, 0);
    k_gemm<<<dim3(gM, 4), 128, 0, stream>>>(ctxbuf, lam_w,     lam_b,     t1,      t2,     MPAD, CDIM, CDIM, 1, 2);
    k_hmean<<<CDIM, 256, 0, stream>>>(ctxbuf, t2, orders, wts, hmeanb);
    k_proj<<<1, 256, 0, stream>>>(hmeanb, omega, projb);
    k_final<<<1, 352, 0, stream>>>(ybuf, projb, ca_w, ca_b, outp);
}
